// _LAKernel_64450279244021
// MI455X (gfx1250) — matrix operand order repaired and checked by reading, not compiled, not run on hardware
//
#include <hip/hip_runtime.h>

// CDNA5 / gfx1250 chunked linear attention scan.
// One block per (batch,head); 16 waves; wave (wm,wn) owns the 16x16 tile of
// every 64x64 product; KV^T state lives in fp32 accumulator VGPRs across the
// 64-chunk scan. v_wmma_f32_16x16x32_f16 for all GEMMs.
// LDS rows padded to 72 halves (36 dwords, gcd(36,64)=4) -> conflict-free
// b128 fragment loads. Global loads double-buffered to hide HBM latency.
// Output stores: one base address per lane + immediate offsets (clause-able).

typedef _Float16 v16h __attribute__((ext_vector_type(16)));
typedef _Float16 v8h  __attribute__((ext_vector_type(8)));
typedef float    v8f  __attribute__((ext_vector_type(8)));
typedef float    v4f  __attribute__((ext_vector_type(4)));

#define D    64      // head dim
#define BS   64      // chunk size
#define TCH  64      // chunks (4096/64)
#define LDH  72      // padded LDS row stride in halves (144B, 16B-aligned)

// A-fragment (16x32 f16 MxK): lane m = lane&15, khalf = lane>>4.
// e -> K = kbase + 16*khalf + e  => two 16B reads.
__device__ __forceinline__ v16h load_a_frag(const _Float16* p) {
  v8h lo = *(const v8h*)(p);
  v8h hi = *(const v8h*)(p + 8);
  v16h a;
#pragma unroll
  for (int e = 0; e < 8; ++e) { a[e] = lo[e]; a[e + 8] = hi[e]; }
  return a;
}

// B-fragment (32x16 f16 KxN) read from a B^T-row-major array:
// lane n = lane&15, khalf = lane>>4, e -> K = kbase + 16*khalf + e.
__device__ __forceinline__ v16h load_b_frag(const _Float16* p) {
  v8h lo = *(const v8h*)(p);
  v8h hi = *(const v8h*)(p + 8);
  v16h b;
#pragma unroll
  for (int e = 0; e < 8; ++e) { b[e] = lo[e]; b[e + 8] = hi[e]; }
  return b;
}

__global__ __launch_bounds__(512, 1)
void la_chunk_scan(const float* __restrict__ q, const float* __restrict__ k,
                   const float* __restrict__ v, float* __restrict__ out) {
  __shared__ __align__(16) _Float16 sQ  [BS * LDH];  // Q   row-major  (A for S and Q@KV)
  __shared__ __align__(16) _Float16 sK  [BS * LDH];  // K   row-major  (B^T for S)
  __shared__ __align__(16) _Float16 sKt [D  * LDH];  // K^T            (B^T for KV^T update)
  __shared__ __align__(16) _Float16 sVt [D  * LDH];  // V^T            (A for KV^T, B^T for S@V)
  __shared__ __align__(16) _Float16 sS  [BS * LDH];  // masked scores  (A for S@V)
  __shared__ __align__(16) _Float16 sKVT[D  * LDH];  // KV^T f16 copy  (B^T for Q@KV)

  const int tid   = threadIdx.x;
  const int lane  = tid & 31;
  const int wave  = tid >> 5;
  const int wm    = wave >> 2;          // row tile 0..3 (wave-uniform)
  const int wn    = wave & 3;           // col tile 0..3
  const int khalf = lane >> 4;
  const int ln16  = lane & 15;

  const size_t bh_base = (size_t)blockIdx.x * (size_t)(TCH * BS * D);

  v8f kvT_acc = {};   // KV^T tile: rows d2 = wm*16.., cols d1 = wn*16.. (fp32)

  // Q loader: 8 contiguous floats per thread (2 x global b128)
  const int qflat = tid * 8;
  const int qrow  = qflat >> 6;
  const int qcol  = qflat & 63;
  // K/V loader: column strip -> column cidx, rows r0..r0+7 (8 x b32, 128B/wave coalesced)
  const int cidx  = tid & 63;
  const int r0    = (tid >> 6) * 8;

  // per-lane C/D tile position (invariant): row0 = wm*16 + 8*khalf, col = wn*16 + ln16
  const int crow0   = wm * 16 + 8 * khalf;
  const int ccol    = wn * 16 + ln16;
  // single output base pointer for this lane; per-chunk advance is +BS*D floats
  float* po = out + bh_base + (size_t)(crow0 * D + ccol);

  // ---- prologue: prefetch chunk 0 into registers ----
  v4f   q0 = *(const v4f*)(q + bh_base + qflat);
  v4f   q1 = *(const v4f*)(q + bh_base + qflat + 4);
  float kr[8], vr[8];
#pragma unroll
  for (int i = 0; i < 8; ++i) {
    kr[i] = k[bh_base + (size_t)(r0 + i) * D + cidx];
    vr[i] = v[bh_base + (size_t)(r0 + i) * D + cidx];
  }

#pragma unroll 1
  for (int t = 0; t < TCH; ++t) {
    // ---- publish KV^T state (f16, row-major => lane-contiguous stores) ----
#pragma unroll
    for (int r = 0; r < 8; ++r)
      sKVT[(crow0 + r) * LDH + ccol] = (_Float16)kvT_acc[r];

    // ---- stage chunk t (registers -> f16 LDS) ----
    {
      v8h qh;
#pragma unroll
      for (int i = 0; i < 4; ++i) { qh[i] = (_Float16)q0[i]; qh[4 + i] = (_Float16)q1[i]; }
      *(v8h*)&sQ[qrow * LDH + qcol] = qh;
      v8h kh, vh;
#pragma unroll
      for (int i = 0; i < 8; ++i) {
        kh[i] = (_Float16)kr[i];
        vh[i] = (_Float16)vr[i];
        sK[(r0 + i) * LDH + cidx] = kh[i];   // lane-contiguous b16 stores
      }
      *(v8h*)&sKt[cidx * LDH + r0] = kh;     // one contiguous b128 store
      *(v8h*)&sVt[cidx * LDH + r0] = vh;     // one contiguous b128 store
    }

    // ---- prefetch chunk t+1 (in flight across barrier + all WMMA work) ----
    if (t + 1 < TCH) {
      const size_t nbase = bh_base + (size_t)(t + 1) * (BS * D);
      q0 = *(const v4f*)(q + nbase + qflat);
      q1 = *(const v4f*)(q + nbase + qflat + 4);
#pragma unroll
      for (int i = 0; i < 8; ++i) {
        kr[i] = k[nbase + (size_t)(r0 + i) * D + cidx];
        vr[i] = v[nbase + (size_t)(r0 + i) * D + cidx];
      }
    }
    __syncthreads();

    // ---- S = tril(Q K^T), tile (wm,wn); wm<wn tiles identically zero ----
    if (wm >= wn) {                       // wave-uniform: EXEC stays all-1
      v8f s_acc = {};
#pragma unroll
      for (int ks = 0; ks < 2; ++ks) {
        v16h a = load_a_frag(&sQ[(wm * 16 + ln16) * LDH + ks * 32 + 16 * khalf]);
        v16h b = load_b_frag(&sK[(wn * 16 + ln16) * LDH + ks * 32 + 16 * khalf]);
        s_acc = __builtin_amdgcn_wmma_f32_16x16x32_f16(false, a, false, b,
                                                       (short)0, s_acc, false, false);
      }
#pragma unroll
      for (int r = 0; r < 8; ++r) {
        // causal mask: keep j <= i
        sS[(crow0 + r) * LDH + ccol] = (_Float16)((ccol <= crow0 + r) ? s_acc[r] : 0.0f);
      }
    } else {
#pragma unroll
      for (int r = 0; r < 8; ++r)
        sS[(crow0 + r) * LDH + ccol] = (_Float16)0.0f;
    }
    __syncthreads();

    // ---- O = S @ V + Q @ KV, tile (wm,wn) ----
    v8f o_acc = {};
    const int ksmax = (wm >= 2) ? 2 : 1;        // S(:,j)==0 for j > wm*16+15
    for (int ks = 0; ks < ksmax; ++ks) {
      v16h a = load_a_frag(&sS[(wm * 16 + ln16) * LDH + ks * 32 + 16 * khalf]);
      v16h b = load_b_frag(&sVt[(wn * 16 + ln16) * LDH + ks * 32 + 16 * khalf]);
      o_acc = __builtin_amdgcn_wmma_f32_16x16x32_f16(false, a, false, b,
                                                     (short)0, o_acc, false, false);
    }
#pragma unroll
    for (int ks = 0; ks < 2; ++ks) {
      v16h a = load_a_frag(&sQ[(wm * 16 + ln16) * LDH + ks * 32 + 16 * khalf]);
      v16h b = load_b_frag(&sKVT[(wn * 16 + ln16) * LDH + ks * 32 + 16 * khalf]);
      o_acc = __builtin_amdgcn_wmma_f32_16x16x32_f16(false, a, false, b,
                                                     (short)0, o_acc, false, false);
    }

    // ---- KV^T += V^T @ K  (independent WMMAs: fill the WMMA->VALU hazard
    //      window before o_acc is read by the stores below) ----
#pragma unroll
    for (int ks = 0; ks < 2; ++ks) {
      v16h a = load_a_frag(&sVt[(wm * 16 + ln16) * LDH + ks * 32 + 16 * khalf]);
      v16h b = load_b_frag(&sKt[(wn * 16 + ln16) * LDH + ks * 32 + 16 * khalf]);
      kvT_acc = __builtin_amdgcn_wmma_f32_16x16x32_f16(false, a, false, b,
                                                       (short)0, kvT_acc, false, false);
    }

    // ---- store O tile: one base address, immediate row offsets (clause) ----
#pragma unroll
    for (int r = 0; r < 8; ++r)
      po[r * D] = o_acc[r];
    po += BS * D;

    __syncthreads();   // protect LDS before next chunk overwrites it
  }
}

extern "C" void kernel_launch(void* const* d_in, const int* in_sizes, int n_in,
                              void* d_out, int out_size, void* d_ws, size_t ws_size,
                              hipStream_t stream) {
  (void)n_in; (void)out_size; (void)d_ws; (void)ws_size;
  const float* q = (const float*)d_in[0];
  const float* k = (const float*)d_in[1];
  const float* v = (const float*)d_in[2];
  float* out = (float*)d_out;

  const int bh = in_sizes[0] / (4096 * 64);   // B*H blocks (seq=4096, d=64)
  hipLaunchKernelGGL(la_chunk_scan, dim3(bh), dim3(512), 0, stream, q, k, v, out);
}
